// NL_74234214744691
// MI455X (gfx1250) — 16-bit and 8-bit matrix operand order repaired and checked by reading, compiled, not run on hardware
//
#include <hip/hip_runtime.h>
#include <hip/hip_bf16.h>

constexpr int Bb   = 4;
constexpr int Cin  = 256;
constexpr int Cout = 128;
constexpr int Nn   = 4096;
constexpr float EPSF = 1e-5f;

typedef __attribute__((ext_vector_type(16))) __bf16       v16bf;
typedef __attribute__((ext_vector_type(16))) int          v16i;
typedef __attribute__((ext_vector_type(8)))  float        v8f;
typedef __attribute__((ext_vector_type(4)))  unsigned int uint4v;
typedef __attribute__((ext_vector_type(2)))  unsigned int uint2v;

union Frag16 { v16bf v; uint4v u[2]; };
union Frag8  { v16i  v; uint4v u[4]; };

__device__ inline v8f wmma_bf16(v16bf a, v16bf b, v8f c) {
  return __builtin_amdgcn_wmma_f32_16x16x32_bf16(
      false, a, false, b, (short)0, c, false, false);
}
__device__ inline v8f wmma_fp8(v16i a, v16i b, v8f c) {
  return __builtin_amdgcn_wmma_f32_16x16x128_fp8_fp8(a, b, (short)0, c, false, false);
}

__device__ inline v16bf load_a(const unsigned short* __restrict__ row, int k0, int hi) {
  Frag16 f;
  f.u[0] = *reinterpret_cast<const uint4v*>(row + k0 + 8 * hi);
  f.u[1] = *reinterpret_cast<const uint4v*>(row + k0 + 16 + 8 * hi);
  return f.v;
}
__device__ inline v16bf load_b(const unsigned short* __restrict__ row, int k0, int hi) {
  Frag16 f;
  f.u[0] = *reinterpret_cast<const uint4v*>(row + k0 + 8 * hi);
  f.u[1] = *reinterpret_cast<const uint4v*>(row + k0 + 16 + 8 * hi);
  return f.v;
}
__device__ inline v16i load_a8(const unsigned char* __restrict__ laneBase) {
  Frag8 f;
  #pragma unroll
  for (int j = 0; j < 4; ++j)
    f.u[j] = *reinterpret_cast<const uint4v*>(laneBase + 16 * j);
  return f.v;
}
__device__ inline v16i load_b8(const unsigned char* __restrict__ row, int k0, int hi) {
  Frag8 f;
  uint2v* d = reinterpret_cast<uint2v*>(&f);
  #pragma unroll
  for (int c = 0; c < 8; ++c)
    d[c] = *reinterpret_cast<const uint2v*>(row + k0 + 16 * c + 8 * hi);
  return f.v;
}

__device__ inline unsigned short f2bf(float x) {
  unsigned int u = __float_as_uint(x);
  u = (u + 0x7FFFu + ((u >> 16) & 1u)) >> 16;
  return (unsigned short)u;
}
__device__ inline float bf2f(unsigned short s) {
  return __uint_as_float(((unsigned int)s) << 16);
}
__device__ inline unsigned pack2(float a, float b) {
  return (unsigned)f2bf(a) | ((unsigned)f2bf(b) << 16);
}
__device__ inline unsigned f2fp8(float x) {
  unsigned u   = __float_as_uint(x);
  unsigned sgn = (u >> 24) & 0x80u;
  unsigned mag = u & 0x7FFFFFFFu;
  if (mag >= 0x43E00000u) return sgn | 0x7Eu;
  if (mag < 0x3C800000u) {
    unsigned q = (unsigned)rintf(__uint_as_float(mag) * 512.f);
    return sgn | q;
  }
  unsigned r = mag + 0x0007FFFFu + ((mag >> 20) & 1u);
  if (r >= 0x43E00000u) return sgn | 0x7Eu;
  unsigned e8 = (r >> 23) - 120u;
  unsigned m3 = (r >> 20) & 7u;
  return sgn | (e8 << 3) | m3;
}
__device__ inline int swz128(int n) {
  const int w = n & 127;
  return (n & ~127) | ((w & 8) << 3) | ((w & 0x70) >> 1) | (w & 7);
}

__global__ __launch_bounds__(256) void k_transpose_x(const float* __restrict__ x,
                                                     unsigned short* __restrict__ xT) {
  __shared__ float tile[32][33];
  const int b  = blockIdx.z;
  const int c0 = blockIdx.y * 32;
  const int n0 = blockIdx.x * 32;
  const int tx = threadIdx.x, ty = threadIdx.y;
  const float* xb = x + (size_t)b * Cin * Nn;
  #pragma unroll
  for (int i = 0; i < 32; i += 8)
    tile[ty + i][tx] = xb[(size_t)(c0 + ty + i) * Nn + n0 + tx];
  __syncthreads();
  unsigned short* xtb = xT + (size_t)b * Nn * Cin;
  #pragma unroll
  for (int i = 0; i < 32; i += 8)
    xtb[(size_t)(n0 + ty + i) * Cin + c0 + tx] = f2bf(tile[tx][ty + i]);
}

__global__ __launch_bounds__(256) void k_prep(
    const float* __restrict__ Wq, const float* __restrict__ Wk, const float* __restrict__ Wv,
    const float* __restrict__ Ww, const float* __restrict__ bw,
    const float* __restrict__ gamma, const float* __restrict__ beta,
    const float* __restrict__ mean, const float* __restrict__ var,
    unsigned short* __restrict__ Wqb, unsigned short* __restrict__ Wkb,
    unsigned short* __restrict__ Wvb, unsigned short* __restrict__ Wob,
    float* __restrict__ bprime) {
  const int i = blockIdx.x * 256 + threadIdx.x;
  if (i < Cout * Cin) {
    Wqb[i] = f2bf(Wq[i]); Wkb[i] = f2bf(Wk[i]); Wvb[i] = f2bf(Wv[i]);
  }
  if (i < Cin * Cout) {
    const int o = i / Cout;
    const float inv = gamma[o] * rsqrtf(var[o] + EPSF);
    Wob[i] = f2bf(Ww[i] * inv);
  }
  if (i < Cin) {
    const float inv = gamma[i] * rsqrtf(var[i] + EPSF);
    bprime[i] = bw[i] * inv + beta[i] - mean[i] * inv;
  }
}

__global__ __launch_bounds__(256) void k_qkv(
    const unsigned short* __restrict__ xT,
    const unsigned short* __restrict__ Wqb, const unsigned short* __restrict__ Wkb,
    const unsigned short* __restrict__ Wvb,
    const float* __restrict__ bq, const float* __restrict__ bk, const float* __restrict__ bv,
    unsigned short* __restrict__ qT, unsigned short* __restrict__ kT,
    unsigned char* __restrict__ vswz) {
  const int b    = blockIdx.y;
  const int nb   = blockIdx.x * 16;
  const int lane = threadIdx.x & 31;
  const int wave = threadIdx.x >> 5;
  const int hi   = lane >> 4;
  const int lx   = lane & 15;
  const int n    = nb + lx;
  const int ob   = wave * 16;

  const unsigned short* xrow = xT  + ((size_t)b * Nn + n) * Cin;
  const unsigned short* aq   = Wqb + (size_t)(ob + lx) * Cin;
  const unsigned short* ak   = Wkb + (size_t)(ob + lx) * Cin;
  const unsigned short* av   = Wvb + (size_t)(ob + lx) * Cin;

  v8f cq = {}; v8f ck = {}; v8f cv = {};
  #pragma unroll
  for (int k0 = 0; k0 < Cin; k0 += 32) {
    v16bf bfr = load_b(xrow, k0, hi);
    cq = wmma_bf16(load_a(aq, k0, hi), bfr, cq);
    ck = wmma_bf16(load_a(ak, k0, hi), bfr, ck);
    cv = wmma_bf16(load_a(av, k0, hi), bfr, cv);
  }
  const int o0 = ob + 8 * hi;
  {
    uint4v p;
    #pragma unroll
    for (int r = 0; r < 4; ++r)
      p[r] = pack2(cq[2 * r] + bq[o0 + 2 * r], cq[2 * r + 1] + bq[o0 + 2 * r + 1]);
    *reinterpret_cast<uint4v*>(qT + ((size_t)b * Nn + n) * Cout + o0) = p;
  }
  {
    uint4v p;
    #pragma unroll
    for (int r = 0; r < 4; ++r)
      p[r] = pack2(ck[2 * r] + bk[o0 + 2 * r], ck[2 * r + 1] + bk[o0 + 2 * r + 1]);
    *reinterpret_cast<uint4v*>(kT + ((size_t)b * Nn + n) * Cout + o0) = p;
  }
  const int ns = swz128(n);
  #pragma unroll
  for (int r = 0; r < 8; ++r) {
    const int o = o0 + r;
    vswz[((size_t)b * Cout + o) * Nn + ns] = (unsigned char)f2fp8(cv[r] + bv[o]);
  }
}

__global__ __launch_bounds__(256) void k_attn(
    const unsigned short* __restrict__ qT, const unsigned short* __restrict__ kT,
    unsigned char* __restrict__ Em) {
  __shared__ float s_red[8][16];
  __shared__ float s_fin[16];
  const int b    = blockIdx.y;
  const int nb   = blockIdx.x * 16;
  const int lane = threadIdx.x & 31;
  const int wave = threadIdx.x >> 5;
  const int hi   = lane >> 4;
  const int lx   = lane & 15;
  const int n    = nb + lx;

  const unsigned short* krow = kT + ((size_t)b * Nn + n) * Cout;
  v16bf afr[4];
  #pragma unroll
  for (int c = 0; c < 4; ++c) afr[c] = load_a(krow, c * 32, hi);

  float lsum[8];
  #pragma unroll
  for (int r = 0; r < 8; ++r) lsum[r] = 0.f;

  for (int mt = wave; mt < Nn / 16; mt += 8) {
    const int m = mt * 16 + lx;
    const unsigned short* qrow = qT + ((size_t)b * Nn + m) * Cout;
    v8f acc = {};
    #pragma unroll
    for (int c = 0; c < 4; ++c) acc = wmma_bf16(afr[c], load_b(qrow, c * 32, hi), acc);
    #pragma unroll
    for (int r = 0; r < 8; ++r) lsum[r] += __expf(fminf(acc[r], 80.f));
  }
  #pragma unroll
  for (int mask = 1; mask < 16; mask <<= 1)
    #pragma unroll
    for (int r = 0; r < 8; ++r) lsum[r] += __shfl_xor(lsum[r], mask, 32);
  if (lx == 0) {
    #pragma unroll
    for (int r = 0; r < 8; ++r) s_red[wave][8 * hi + r] = lsum[r];
  }
  __syncthreads();
  if (threadIdx.x < 16) {
    float s = 0.f;
    #pragma unroll
    for (int w = 0; w < 8; ++w) s += s_red[w][threadIdx.x];
    s_fin[threadIdx.x] = s;
  }
  __syncthreads();
  float scl[8];
  #pragma unroll
  for (int r = 0; r < 8; ++r) scl[r] = 448.f / s_fin[8 * hi + r];

  for (int mt = wave; mt < Nn / 16; mt += 8) {
    const int m = mt * 16 + lx;
    const unsigned short* qrow = qT + ((size_t)b * Nn + m) * Cout;
    v8f acc = {};
    #pragma unroll
    for (int c = 0; c < 4; ++c) acc = wmma_bf16(afr[c], load_b(qrow, c * 32, hi), acc);
    uint2v p; p[0] = 0u; p[1] = 0u;
    #pragma unroll
    for (int r = 0; r < 4; ++r)
      p[0] |= f2fp8(__expf(fminf(acc[r], 80.f)) * scl[r]) << (8 * r);
    #pragma unroll
    for (int r = 4; r < 8; ++r)
      p[1] |= f2fp8(__expf(fminf(acc[r], 80.f)) * scl[r]) << (8 * (r - 4));
    *reinterpret_cast<uint2v*>(Em + ((size_t)b * Nn + m) * Nn + nb + 8 * hi) = p;
  }
}

__global__ __launch_bounds__(256) void k_ygemm(
    const unsigned char* __restrict__ vswz, const unsigned char* __restrict__ Em,
    unsigned short* __restrict__ yT) {
  const int b    = blockIdx.y;
  const int mb   = blockIdx.x * 16;
  const int lane = threadIdx.x & 31;
  const int wave = threadIdx.x >> 5;
  const int hi   = lane >> 4;
  const int lx   = lane & 15;
  const int m    = mb + lx;
  const int c0   = wave * 16;

  const unsigned char* arow = vswz + ((size_t)b * Cout + c0 + lx) * Nn + 64 * hi;
  const unsigned char* brow = Em   + ((size_t)b * Nn + m) * Nn;
  v8f acc = {};
  #pragma unroll 2
  for (int k0 = 0; k0 < Nn; k0 += 128)
    acc = wmma_fp8(load_a8(arow + k0), load_b8(brow, k0, hi), acc);

  constexpr float inv448 = 1.0f / 448.0f;
  uint4v p;
  #pragma unroll
  for (int r = 0; r < 4; ++r)
    p[r] = pack2(acc[2 * r] * inv448, acc[2 * r + 1] * inv448);
  *reinterpret_cast<uint4v*>(yT + ((size_t)b * Nn + m) * Cout + c0 + 8 * hi) = p;
}

__global__ __launch_bounds__(256) void k_out(
    const unsigned short* __restrict__ yT, const unsigned short* __restrict__ Wob,
    const float* __restrict__ bprime, const float* __restrict__ x,
    float* __restrict__ out) {
  const int b    = blockIdx.y;
  const int nb   = blockIdx.x * 16;
  const int lane = threadIdx.x & 31;
  const int wave = threadIdx.x >> 5;
  const int hi   = lane >> 4;
  const int lx   = lane & 15;
  const int n    = nb + lx;

  const unsigned short* brow = yT + ((size_t)b * Nn + n) * Cout;
  v16bf bfr[4];
  #pragma unroll
  for (int c = 0; c < 4; ++c) bfr[c] = load_b(brow, c * 32, hi);

  for (int t = wave; t < Cin / 16; t += 8) {
    const unsigned short* arow = Wob + (size_t)(t * 16 + lx) * Cout;
    v8f acc = {};
    #pragma unroll
    for (int c = 0; c < 4; ++c) acc = wmma_bf16(load_a(arow, c * 32, hi), bfr[c], acc);
    const int o0 = t * 16 + 8 * hi;
    #pragma unroll
    for (int r = 0; r < 8; ++r) {
      const int o = o0 + r;
      const size_t idx = ((size_t)b * Cin + o) * Nn + n;
      out[idx] = acc[r] + bprime[o] + x[idx];
    }
  }
}

constexpr size_t SZ_XT   = (size_t)Bb * Nn * Cin * 2;
constexpr size_t SZ_QT   = (size_t)Bb * Nn * Cout * 2;
constexpr size_t SZ_KT   = SZ_QT;
constexpr size_t SZ_V    = (size_t)Bb * Cout * Nn;
constexpr size_t SZ_YT   = SZ_QT;
constexpr size_t SZ_WQB  = (size_t)Cout * Cin * 2;
constexpr size_t SZ_WOB  = (size_t)Cin * Cout * 2;
constexpr size_t SZ_BP   = (size_t)Cin * 4;
constexpr size_t SZ_EM   = (size_t)Bb * Nn * Nn;

extern "C" void kernel_launch(void* const* d_in, const int* in_sizes, int n_in,
                              void* d_out, int out_size, void* d_ws, size_t ws_size,
                              hipStream_t stream) {
  const float* x     = (const float*)d_in[0];
  const float* Wq    = (const float*)d_in[1];
  const float* bq    = (const float*)d_in[2];
  const float* Wk    = (const float*)d_in[3];
  const float* bk    = (const float*)d_in[4];
  const float* Wv    = (const float*)d_in[5];
  const float* bv    = (const float*)d_in[6];
  const float* Ww    = (const float*)d_in[7];
  const float* bw    = (const float*)d_in[8];
  const float* gamma = (const float*)d_in[9];
  const float* beta  = (const float*)d_in[10];
  const float* mean  = (const float*)d_in[11];
  const float* var   = (const float*)d_in[12];
  float* out = (float*)d_out;

  unsigned char* w = (unsigned char*)d_ws;
  size_t off = 0;
  unsigned short* xT   = (unsigned short*)(w + off); off += SZ_XT;
  unsigned short* qT   = (unsigned short*)(w + off); off += SZ_QT;
  unsigned short* kT   = (unsigned short*)(w + off); off += SZ_KT;
  unsigned char*  vswz = (unsigned char*)(w + off);  off += SZ_V;
  unsigned short* yT   = (unsigned short*)(w + off); off += SZ_YT;
  unsigned short* Wqb  = (unsigned short*)(w + off); off += SZ_WQB;
  unsigned short* Wkb  = (unsigned short*)(w + off); off += SZ_WQB;
  unsigned short* Wvb  = (unsigned short*)(w + off); off += SZ_WQB;
  unsigned short* Wob  = (unsigned short*)(w + off); off += SZ_WOB;
  float*          bpr  = (float*)(w + off);          off += SZ_BP;
  unsigned char*  Em   = (unsigned char*)(w + off);  off += SZ_EM;
  if (ws_size < off) return;

  k_prep<<<dim3((Cout * Cin + 255) / 256), dim3(256), 0, stream>>>(
      Wq, Wk, Wv, Ww, bw, gamma, beta, mean, var, Wqb, Wkb, Wvb, Wob, bpr);
  k_transpose_x<<<dim3(Nn / 32, Cin / 32, Bb), dim3(32, 8), 0, stream>>>(x, xT);
  k_qkv<<<dim3(Nn / 16, Bb), dim3(256), 0, stream>>>(
      xT, Wqb, Wkb, Wvb, bq, bk, bv, qT, kT, vswz);
  k_attn<<<dim3(Nn / 16, Bb), dim3(256), 0, stream>>>(qT, kT, Em);
  k_ygemm<<<dim3(Nn / 16, Bb), dim3(256), 0, stream>>>(vswz, Em, yT);
  k_out<<<dim3(Nn / 16, Bb), dim3(256), 0, stream>>>(yT, Wob, bpr, x, out);
}
